// KANLayer_17265768530645
// MI455X (gfx1250) — matrix operand order repaired and checked by reading, not compiled, not run on hardware
//
#include <hip/hip_runtime.h>

// ---------------------------------------------------------------------------
// KAN layer on MI455X (gfx1250).
//   y[b,o] = sum_{i,k} hat_k(pos[b,i]) * W[o,i,k] + bias[o] + sum_i x[b,i]*S[o,i]
// hat_k(p) = max(0, 1-|p-k|) reproduces the reference interpolation exactly,
// so the spline term is a dense GEMM over ik (8192) whose A-fragments are
// generated on the fly in registers -> v_wmma_f32_16x16x32_bf16.
// Skip GEMM fused into the same loop (x read once); x split hi+lo bf16.
// W tiles staged via TDM (tensor_load_to_lds, 6-arg form) double-buffered;
// skip_w tiles via global_load_async_to_lds_b128 (typed-pointer form).
// HBM ~26 MB -> ~1.1 us floor @ 23.3 TB/s; 17.2 GFLOP of bf16 WMMA.
// ---------------------------------------------------------------------------

typedef __bf16 bf16;
typedef __attribute__((ext_vector_type(16))) __bf16 v16bf;
typedef __attribute__((ext_vector_type(8)))  float  v8f;
typedef __attribute__((ext_vector_type(4)))  unsigned int u32x4;
typedef __attribute__((ext_vector_type(8)))  int i32x8;
typedef __attribute__((ext_vector_type(4)))  int i32x4;

static constexpr int BROWS = 8192;
static constexpr int DIN   = 512;
static constexpr int DOUT  = 128;
static constexpr int KK    = 16;

static constexpr int BLOCK_M  = 128;  // 8 waves x 16 rows
static constexpr int NTHREADS = 256;

#if defined(__has_builtin)
#if __has_builtin(__builtin_amdgcn_tensor_load_to_lds) && __has_builtin(__builtin_amdgcn_s_wait_tensorcnt)
#define HAVE_TDM 1
#endif
#if __has_builtin(__builtin_amdgcn_global_load_async_to_lds_b128) && __has_builtin(__builtin_amdgcn_s_wait_asynccnt)
#define HAVE_ASYNC 1
#endif
#endif

// ---------------- preprocessing: f32 -> bf16 into workspace ----------------
__global__ __launch_bounds__(256) void kan_convert(
    const float* __restrict__ w, const float* __restrict__ sw,
    bf16* __restrict__ wb, bf16* __restrict__ sb)
{
  const size_t NW = (size_t)DOUT * DIN * KK;
  const size_t NS = (size_t)DOUT * DIN;
  const size_t stride = (size_t)gridDim.x * blockDim.x;
  for (size_t t = (size_t)blockIdx.x * blockDim.x + threadIdx.x; t < NW; t += stride)
    wb[t] = (bf16)w[t];
  for (size_t t = (size_t)blockIdx.x * blockDim.x + threadIdx.x; t < NS; t += stride)
    sb[t] = (bf16)sw[t];
}

// ------------------------------- main kernel -------------------------------
__global__ __launch_bounds__(NTHREADS) void kan_gemm(
    const float* __restrict__ x,     // [8192][512]
    const bf16*  __restrict__ wb,    // [128][512][16] bf16
    const bf16*  __restrict__ sb,    // [128][512]     bf16
    const float* __restrict__ bias,  // [128]
    float*       __restrict__ out)   // [8192][128]
{
  // LDS carve (60 KB):
  //   Wbuf  [2][4][128][16] bf16 : b*16384 + ii*4096 + o*32 bytes (TDM dest)
  //   pos   [2][128][4] f32
  //   xh,xl [128][32] bf16 ring (one 32-i group)
  //   swt   [128][32] bf16
  __shared__ __align__(32) unsigned char smem[61440];
  bf16*  Wbuf   = (bf16*)(smem);
  float* posbuf = (float*)(smem + 32768);
  bf16*  xh     = (bf16*)(smem + 36864);
  bf16*  xl     = (bf16*)(smem + 45056);
  bf16*  swt    = (bf16*)(smem + 53248);

  const int tid  = threadIdx.x;
  const int lane = tid & 31;
  const int wave = tid >> 5;
  const int half = lane >> 4;
  const int l16  = lane & 15;
  const int blockRow = blockIdx.x * BLOCK_M;

  v8f acc[8];
  #pragma unroll
  for (int nt = 0; nt < 8; nt++)
    #pragma unroll
    for (int e = 0; e < 8; e++) acc[nt][e] = 0.0f;

  const float inv_range = 1.0f / (6.0f + 1e-8f);

  // ---- stage pos (for stage t, buffer t&1) + x hi/lo ring (cols (t&7)*4) ----
  auto stage_x = [&](int t) {
    if (tid < 128) {
      const int r = tid;
      const float4 xv = *(const float4*)&x[(size_t)(blockRow + r) * DIN + t * 4];
      float* pb = posbuf + (t & 1) * 512 + r * 4;
      float p[4]; const float v[4] = {xv.x, xv.y, xv.z, xv.w};
      unsigned hq[2], lq[2];
      unsigned short hs[4], ls[4];
      #pragma unroll
      for (int q = 0; q < 4; q++) {
        float tt = (v[q] + 3.0f) * inv_range;
        tt = fminf(fmaxf(tt, 0.0f), 1.0f);
        p[q] = tt * 15.0f;
        bf16 h = (bf16)v[q];
        bf16 l = (bf16)(v[q] - (float)h);
        hs[q] = __builtin_bit_cast(unsigned short, h);
        ls[q] = __builtin_bit_cast(unsigned short, l);
      }
      *(float4*)pb = make_float4(p[0], p[1], p[2], p[3]);
      hq[0] = hs[0] | ((unsigned)hs[1] << 16); hq[1] = hs[2] | ((unsigned)hs[3] << 16);
      lq[0] = ls[0] | ((unsigned)ls[1] << 16); lq[1] = ls[2] | ((unsigned)ls[3] << 16);
      const int cb = (t & 7) * 4;
      *(uint2*)&xh[r * 32 + cb] = make_uint2(hq[0], hq[1]);
      *(uint2*)&xl[r * 32 + cb] = make_uint2(lq[0], lq[1]);
    }
  };

  // ---- W tile for stage t into Wbuf[t&1]: 4 x (2D TDM: 32B x 128 rows) ----
  auto issue_w = [&](int t) {
#ifdef HAVE_TDM
    if (wave == 0) {
      const unsigned ldsbase =
          (unsigned)(unsigned long long)Wbuf + (unsigned)((t & 1) * 16384);
      const unsigned long long gbase =
          (unsigned long long)wb + (unsigned long long)t * 128ull;
      #pragma unroll
      for (int ii = 0; ii < 4; ii++) {
        const unsigned long long ga = gbase + (unsigned long long)ii * 32ull;
        u32x4 g0;
        g0[0] = 1u;                                   // count=1, no gather
        g0[1] = ldsbase + (unsigned)ii * 4096u;       // lds_addr
        g0[2] = (unsigned)ga;                         // global_addr[31:0]
        g0[3] = (unsigned)((ga >> 32) & 0x01FFFFFFu)  // global_addr[56:32]
                | 0x80000000u;                        // type=2 (image)
        i32x8 g1;
        g1[0] = 3 << 16;        // data_size=8B; no pad/iterate/multicast
        g1[1] = 4 << 16;        // tensor_dim0 = 4 units (32B row)
        g1[2] = 128 << 16;      // tensor_dim1 = 128 rows
        g1[3] = 4 << 16;        // tile_dim0 = 4 units
        g1[4] = 128;            // tile_dim1 = 128 (tile_dim2 = 0)
        g1[5] = 2048;           // tensor_dim0_stride = 16KB / 8B
        g1[6] = 0; g1[7] = 0;
        const i32x4 z4 = {0, 0, 0, 0};
        const i32x8 z8 = {0, 0, 0, 0, 0, 0, 0, 0};
        __builtin_amdgcn_tensor_load_to_lds(g0, g1, z4, z4, z8, 0);
      }
    }
#else
    bf16* dst = Wbuf + (t & 1) * 8192;
    for (int c = tid; c < 512; c += NTHREADS) {
      const int o = c >> 2, ii = c & 3;
      const uint4* s4 = (const uint4*)(wb + (size_t)o * (DIN * KK) + (size_t)t * 64 + ii * 16);
      uint4* d4 = (uint4*)(dst + ii * 2048 + o * 16);
      d4[0] = s4[0];
      d4[1] = s4[1];
    }
#endif
  };

  // ---- skip_w tile for i-group g (32 i's) into swt ----
  auto issue_sw = [&](int g) {
#ifdef HAVE_ASYNC
    for (int c = tid; c < 512; c += NTHREADS) {
      const int o = c >> 2, q = c & 3;
      const unsigned long long gp =
          (unsigned long long)sb + (unsigned long long)o * 1024ull +
          (unsigned long long)g * 64ull + (unsigned long long)q * 16ull;
      const unsigned loff =
          (unsigned)(unsigned long long)swt + (unsigned)(o * 64 + q * 16);
      __builtin_amdgcn_global_load_async_to_lds_b128(
          (__attribute__((address_space(1))) i32x4*)gp,
          (__attribute__((address_space(3))) i32x4*)(unsigned long long)loff,
          0, 0);
    }
#else
    for (int c = tid; c < 512; c += NTHREADS) {
      const int o = c >> 2, q = c & 3;
      *(uint4*)(swt + o * 32 + q * 8) =
          *(const uint4*)(sb + (size_t)o * DIN + (size_t)g * 32 + q * 8);
    }
#endif
  };

  // ---- fused skip GEMM over one 32-i group (x = hi + lo bf16) ----
  auto skip_wmma = [&]() {
    const bf16* xrh = xh + (wave * 16 + l16) * 32;
    const bf16* xrl = xl + (wave * 16 + l16) * 32;
    union { v16bf v; uint4 q[2]; } uah, ual;
    uah.q[0] = *(const uint4*)(xrh + 8 * half);
    uah.q[1] = *(const uint4*)(xrh + 16 + 8 * half);
    ual.q[0] = *(const uint4*)(xrl + 8 * half);
    ual.q[1] = *(const uint4*)(xrl + 16 + 8 * half);
    const v16bf Ah = uah.v;
    const v16bf Al = ual.v;
    #pragma unroll
    for (int nt = 0; nt < 8; nt++) {
      const int o = nt * 16 + l16;
      union { v16bf v; uint4 q[2]; } ub;
      const bf16* swp = swt + o * 32 + 8 * half;
      ub.q[0] = ((const uint4*)swp)[0];
      ub.q[1] = ((const uint4*)swp)[2];
      acc[nt] = __builtin_amdgcn_wmma_f32_16x16x32_bf16(
          false, Ah, false, ub.v, (short)0, acc[nt], false, false);
      acc[nt] = __builtin_amdgcn_wmma_f32_16x16x32_bf16(
          false, Al, false, ub.v, (short)0, acc[nt], false, false);
    }
  };

  // =========================== pipelined main loop ===========================
  stage_x(0);
  issue_w(0);

  for (int s = 0; s < 128; s++) {
#ifdef HAVE_TDM
    if (wave == 0) __builtin_amdgcn_s_wait_tensorcnt((short)0);
#endif
#ifdef HAVE_ASYNC
    if ((s & 7) == 7) __builtin_amdgcn_s_wait_asynccnt((short)0);
#endif
    __syncthreads();   // buf[s&1] + pos[s&1] ready; prev reads done

    if ((s & 7) == 7) {            // full 32-i group of x staged -> skip GEMM
      skip_wmma();
      __syncthreads();             // reads done before ring is overwritten
    }
    if ((s & 7) == 0) issue_sw(s >> 3);          // skip_w for current group
    if (s + 1 < 128) { stage_x(s + 1); issue_w(s + 1); }   // prefetch

    // ---- spline WMMAs for stage s (4 i's = 2 K-chunks of 32) ----
    const bf16*  Wt = Wbuf + (s & 1) * 8192;
    const float* pb = posbuf + (s & 1) * 512;
    #pragma unroll
    for (int cl = 0; cl < 2; cl++) {
      const float2 p01 = *(const float2*)&pb[(wave * 16 + l16) * 4 + 2 * cl];
      v16bf A;
      #pragma unroll
      for (int j = 0; j < 8; j++) {
        const float p  = half ? p01.y : p01.x;
        const float k0 = (float)(2 * j);
        const float d0 = p - k0;
        const float e0 = fmaxf(0.0f, 1.0f - fabsf(d0));
        const float e1 = fmaxf(0.0f, 1.0f - fabsf(d0 - 1.0f));
        A[2 * j]     = (bf16)e0;
        A[2 * j + 1] = (bf16)e1;
      }
      #pragma unroll
      for (int nt = 0; nt < 8; nt++) {
        const int o = nt * 16 + l16;
        const v16bf Bf = *(const v16bf*)(Wt + (2 * cl + half) * 2048 + o * 16);
        acc[nt] = __builtin_amdgcn_wmma_f32_16x16x32_bf16(
            false, A, false, Bf, (short)0, acc[nt], false, false);
      }
    }
  }

  // ============================== epilogue ==============================
  #pragma unroll
  for (int nt = 0; nt < 8; nt++) {
    const int col = nt * 16 + l16;
    const float bv = bias[col];
    #pragma unroll
    for (int v = 0; v < 8; v++) {
      const int row = blockRow + wave * 16 + half * 8 + v;
      out[(size_t)row * DOUT + col] = acc[nt][v] + bv;
    }
  }
}

// ------------------------------- launcher ----------------------------------
extern "C" void kernel_launch(void* const* d_in, const int* in_sizes, int n_in,
                              void* d_out, int out_size, void* d_ws, size_t ws_size,
                              hipStream_t stream) {
  const float* x    = (const float*)d_in[0];  // (8192, 512)
  const float* w    = (const float*)d_in[1];  // (128, 512, 16)
  const float* sw   = (const float*)d_in[2];  // (128, 512)
  const float* bias = (const float*)d_in[3];  // (128,)
  float* out = (float*)d_out;                 // (8192, 128)

  bf16* wb = (bf16*)d_ws;                         // 2 MB
  bf16* sb = wb + (size_t)DOUT * DIN * KK;        // +128 KB

  kan_convert<<<2048, 256, 0, stream>>>(w, sw, wb, sb);
  kan_gemm<<<BROWS / BLOCK_M, NTHREADS, 0, stream>>>(x, wb, sb, bias, out);
}
